// TopologicalLossMaxPool_23175643529661
// MI455X (gfx1250) — matrix operand order repaired and checked by reading, not compiled, not run on hardware
//
#include <hip/hip_runtime.h>
#include <stdint.h>

#define Bn 4
#define Cn 3
#define Hn 256
#define Wn 256
#define HWn 65536
#define BCn 12
#define KCAP 32
#define NSTEP 5
#define OVCHUNK 256

typedef __attribute__((ext_vector_type(16))) _Float16 v16h;
typedef __attribute__((ext_vector_type(8)))  float    v8f;

#ifndef __has_builtin
#define __has_builtin(x) 0
#endif

#if __has_builtin(__builtin_amdgcn_tensor_load_to_lds) && __has_builtin(__builtin_amdgcn_s_wait_tensorcnt)
#define USE_TDM 1
typedef __attribute__((ext_vector_type(4))) unsigned int tdm_g0_t;
typedef __attribute__((ext_vector_type(8))) int          tdm_g1_t;
typedef __attribute__((ext_vector_type(4))) int          tdm_g2_t;
typedef __attribute__((ext_vector_type(8))) int          tdm_g4_t;
#else
#define USE_TDM 0
#endif

#if __has_builtin(__builtin_amdgcn_global_load_async_to_lds_b32)
#define USE_ASYNC_LDS 1
typedef __attribute__((address_space(1))) int gint_t;
typedef __attribute__((address_space(3))) int lint_t;
#else
#define USE_ASYNC_LDS 0
#endif
#if __has_builtin(__builtin_amdgcn_s_wait_asynccnt)
#define HAVE_WAIT_ASYNC 1
#else
#define HAVE_WAIT_ASYNC 0
#endif

// ---------------------------------------------------------------- binarize
// src==0: hard argmax over C of predictions+gumbel (straight-through fwd value)
// src==1: one-hot of integer targets
__global__ void k_binarize(const float* __restrict__ pred,
                           const float* __restrict__ gum,
                           const int* __restrict__ tgt,
                           unsigned char* __restrict__ mask0, int src) {
  int idx = blockIdx.x * blockDim.x + threadIdx.x;
  if (idx >= Bn * HWn) return;
  int b = idx >> 16;
  int p = idx & (HWn - 1);
  int arg = 0;
  if (src == 0) {
    float best = -3.4e38f;
    for (int c = 0; c < Cn; ++c) {
      size_t o = ((size_t)(b * Cn + c)) * HWn + p;
      float y = pred[o] + gum[o];
      if (y > best) { best = y; arg = c; }   // first max, like jnp.argmax
    }
  } else {
    arg = tgt[(size_t)b * HWn + p];
  }
  for (int c = 0; c < Cn; ++c)
    mask0[((size_t)(b * Cn + c)) * HWn + p] = (unsigned char)(c == arg ? 1 : 0);
}

// ---------------------------------------------------------------- morphology
// 5x5 box sum with zero padding; dil==0: erode (sum>=25), dil==1: dilate (sum>=1)
__global__ void k_morph(const unsigned char* __restrict__ in,
                        unsigned char* __restrict__ out, int dil) {
  int idx = blockIdx.x * blockDim.x + threadIdx.x;
  if (idx >= BCn * HWn) return;
  int slot = idx >> 16;
  int p = idx & (HWn - 1);
  int y = p >> 8, x = p & 255;
  int s = 0;
  for (int dy = -2; dy <= 2; ++dy) {
    int yy = y + dy;
    if ((unsigned)yy >= 256u) continue;
    int rb = yy << 8;
    for (int dx = -2; dx <= 2; ++dx) {
      int xx = x + dx;
      if ((unsigned)xx >= 256u) continue;
      s += in[(size_t)slot * HWn + rb + xx];
    }
  }
  out[idx] = (unsigned char)(dil ? (s >= 1) : (s >= 25));
}

// ---------------------------------------------------------------- CC labeling
// Whole 256x256 int32 label image resident in LDS (256KB of the 320KB WGP LDS).
// In-place asynchronous 3x3 max propagation to fixpoint (same fixpoint as Jacobi).
__global__ void __launch_bounds__(1024) k_cc(const unsigned char* __restrict__ mask,
                                             int* __restrict__ Lout) {
  extern __shared__ int Lsh[];          // HWn ints = 256 KB dynamic LDS
  __shared__ int s_changed;
  int slot = blockIdx.x;
  int tid = threadIdx.x;
  unsigned long long mb = 0ull;
  for (int k = 0; k < 64; ++k) {
    int p = tid + (k << 10);
    int m = mask[(size_t)slot * HWn + p];
    Lsh[p] = m ? p : 0;
    if (m) mb |= (1ull << k);
  }
  __syncthreads();
  for (;;) {
    if (tid == 0) s_changed = 0;
    __syncthreads();
    int local = 0;
    for (int k = 0; k < 64; ++k) {
      if (!((mb >> k) & 1ull)) continue;
      int p = tid + (k << 10);
      int y = p >> 8, x = p & 255;
      int v = Lsh[p];
      int nv = v;
      for (int dy = -1; dy <= 1; ++dy) {
        int yy = y + dy;
        if ((unsigned)yy >= 256u) continue;
        int rb = yy << 8;
        for (int dx = -1; dx <= 1; ++dx) {
          int xx = x + dx;
          if ((unsigned)xx >= 256u) continue;
          int w = Lsh[rb + xx];
          nv = w > nv ? w : nv;
        }
      }
      if (nv != v) { Lsh[p] = nv; local = 1; }
    }
    if (local) atomicOr(&s_changed, 1);
    __syncthreads();
    if (!s_changed) break;
  }
  for (int k = 0; k < 64; ++k) {
    int p = tid + (k << 10);
    Lout[(size_t)slot * HWn + p] = Lsh[p];
  }
}

// ---------------------------------------------------------------- sizes / maxL
__global__ void k_sizes_zero(float* __restrict__ sizes, int* __restrict__ maxL) {
  int idx = blockIdx.x * blockDim.x + threadIdx.x;
  if (idx < BCn * HWn) sizes[idx] = 0.f;
  if (idx < BCn) maxL[idx] = 1;
}

__global__ void k_sizes_acc(const int* __restrict__ L, float* __restrict__ sizes,
                            int* __restrict__ maxL) {
  int idx = blockIdx.x * blockDim.x + threadIdx.x;
  if (idx >= BCn * HWn) return;
  int v = L[idx];
  int slot = idx >> 16;
  atomicMax(&maxL[slot], v);
  if (v > 0) atomicAdd(&sizes[((size_t)slot << 16) + v], 1.0f);
}

// ---------------------------------------------------------------- top-32 extraction
// key[p] = (L[p]==p && p>0) ? sizes[p]*p : -1 ; 32 strictly-decreasing selection passes
__global__ void k_extract(const int* __restrict__ L, const float* __restrict__ sizes,
                          const int* __restrict__ maxL,
                          int* __restrict__ labs, float* __restrict__ sval,
                          float* __restrict__ vval, int* __restrict__ validc) {
  __shared__ float sk[256];
  __shared__ int si[256];
  int slot = blockIdx.x, tid = threadIdx.x;
  int mLi = maxL[slot]; if (mLi < 1) mLi = 1;
  float mL = (float)mLi;
  float prevK = 3.4e38f; int prevI = 0x7fffffff;
  for (int sel = 0; sel < KCAP; ++sel) {
    float bk = -2.f; int bi = -1;
    for (int p = tid; p < HWn; p += 256) {
      int v = L[((size_t)slot << 16) + p];
      if (v == p && p > 0) {
        float key = sizes[((size_t)slot << 16) + p] * (float)p;
        bool lt = (key < prevK) || (key == prevK && p < prevI);
        if (lt && (key > bk || (key == bk && p > bi))) { bk = key; bi = p; }
      }
    }
    sk[tid] = bk; si[tid] = bi;
    __syncthreads();
    for (int o = 128; o > 0; o >>= 1) {
      if (tid < o) {
        if (sk[tid + o] > sk[tid] || (sk[tid + o] == sk[tid] && si[tid + o] > si[tid])) {
          sk[tid] = sk[tid + o]; si[tid] = si[tid + o];
        }
      }
      __syncthreads();
    }
    float selK = sk[0]; int selI = si[0];
    __syncthreads();
    if (tid == 0) {
      bool valid = selK > 0.f;
      int lab = valid ? selI : -1;
      float vv = valid ? (float)selI / mL : 0.f;
      labs[slot * KCAP + sel] = lab;
      vval[slot * KCAP + sel] = vv;
      sval[slot * KCAP + sel] = valid ? sizes[((size_t)slot << 16) + selI] * vv : 0.f;
      validc[slot * KCAP + sel] = valid ? 1 : 0;
    }
    prevK = selK; prevI = selI;
  }
}

// ---------------------------------------------------------------- registry reset
__global__ void k_regzero(int* __restrict__ ent_t, int* __restrict__ ent_lab,
                          float* __restrict__ ent_v, int* __restrict__ nbuf,
                          float* __restrict__ bar) {
  int idx = blockIdx.x * blockDim.x + threadIdx.x;
  if (idx < BCn * KCAP * NSTEP) bar[idx] = 0.f;
  if (idx < BCn * KCAP) { ent_t[idx] = 0; ent_lab[idx] = -1; ent_v[idx] = 0.f; }
  if (idx < BCn) nbuf[idx] = 0;
}

// ---------------------------------------------------------------- overlap matmul (WMMA)
// OV[j,k] = v_j * v_k * sum_p 1[Lcur[p]==lab_j] * 1[L_{t_k}[p]==lab_k]
// 32x32 result per slot as 4 f32 16x16 tiles; f16 0/1 operands, f32 accumulate (exact counts).
// 256-pixel tiles of all 5 label maps are staged into LDS by the Tensor Data Mover
// (tensor_load_to_lds + s_wait_tensorcnt) when available; async-b32 / plain fallbacks.
__global__ void __launch_bounds__(32) k_ov(const int* __restrict__ labels,
                                           const int* __restrict__ labs,
                                           const float* __restrict__ vval,
                                           const int* __restrict__ ent_t,
                                           const int* __restrict__ ent_lab,
                                           const float* __restrict__ ent_v,
                                           float* __restrict__ OV, int step) {
  __shared__ int stage[NSTEP * OVCHUNK];    // 5 KB
  int slot = blockIdx.x;
  int lane = threadIdx.x;
  int m = lane & 15;
  bool hi = lane >= 16;
  int labA0 = labs[slot * KCAP + m];
  int labA1 = labs[slot * KCAP + 16 + m];
  int t0 = ent_t[slot * KCAP + m],      lb0 = ent_lab[slot * KCAP + m];
  int t1 = ent_t[slot * KCAP + 16 + m], lb1 = ent_lab[slot * KCAP + 16 + m];
  v8f acc00 = {}, acc01 = {}, acc10 = {}, acc11 = {};
  int koffA = hi ? 8 : 0;
  int koffB = hi ? 16 : 0;
  for (int kb0 = 0; kb0 < HWn; kb0 += OVCHUNK) {
#if USE_TDM
    // TDM descriptor: 2D image type, data_size=4B, tile 256x1 row out of a
    // 65536-element tensor; LDS destination = staging slice for map t.
#pragma unroll
    for (int t = 0; t < NSTEP; ++t) {
      unsigned long long gaddr =
          (unsigned long long)(labels + ((size_t)t * BCn + slot) * HWn + kb0);
      unsigned ldsa = (unsigned)(unsigned long long)&stage[t * OVCHUNK];
      tdm_g0_t g0 = {1u,                                   // count=1 (valid user D#)
                     ldsa,                                 // lds_addr
                     (unsigned)gaddr,                      // global_addr[31:0]
                     (unsigned)(((gaddr >> 32) & 0x01FFFFFFull) | (2u << 30))};
      tdm_g1_t g1 = {(int)(2u << 16),                      // data_size=4B
                     0,                                    // no atomic barrier
                     (int)0x00010001,                      // tensor_dim0=65536, tensor_dim1=1
                     (int)(256u << 16),                    // tile_dim0=256
                     1,                                    // tile_dim1=1
                     65536,                                // tensor_dim0_stride
                     0, 0};
      tdm_g2_t z4 = {0, 0, 0, 0};
      tdm_g4_t z8 = {0, 0, 0, 0, 0, 0, 0, 0};
      __builtin_amdgcn_tensor_load_to_lds(g0, g1, z4, z4, z8, 0);
    }
    __builtin_amdgcn_s_wait_tensorcnt(0);
#elif USE_ASYNC_LDS
#pragma unroll
    for (int t = 0; t < NSTEP; ++t) {
      const int* g = labels + ((size_t)t * BCn + slot) * HWn + kb0 + lane;
#pragma unroll
      for (int q = 0; q < OVCHUNK / 32; ++q) {
        int* l = &stage[t * OVCHUNK + q * 32 + lane];
        __builtin_amdgcn_global_load_async_to_lds_b32(
            (gint_t*)(unsigned long long)(g + q * 32),
            (lint_t*)(unsigned)(unsigned long long)l, 0, 0);
      }
    }
#if HAVE_WAIT_ASYNC
    __builtin_amdgcn_s_wait_asynccnt(0);
#endif
#else
#pragma unroll
    for (int t = 0; t < NSTEP; ++t)
#pragma unroll
      for (int q = 0; q < OVCHUNK / 32; ++q)
        stage[t * OVCHUNK + q * 32 + lane] =
            labels[((size_t)t * BCn + slot) * HWn + kb0 + q * 32 + lane];
#endif
    if (kb0 + OVCHUNK < HWn)   // pull next tile toward the WGP while WMMAs run
      __builtin_prefetch(labels + ((size_t)step * BCn + slot) * HWn + kb0 + OVCHUNK + lane, 0, 1);
    __syncthreads();
#pragma unroll
    for (int c = 0; c < OVCHUNK / 32; ++c) {
      int base = c * 32;
      v16h a0 = {}, a1 = {}, b0 = {}, b1 = {};
#pragma unroll
      for (int i = 0; i < 8; ++i) {
#pragma unroll
        for (int sub = 0; sub < 2; ++sub) {
          int h = 2 * i + sub;
          int krel = ((i < 4) ? (2 * i + sub) : (16 + 2 * (i - 4) + sub)) + koffA;
          int lc = stage[step * OVCHUNK + base + krel];
          a0[h] = (lc == labA0) ? (_Float16)1.0f : (_Float16)0.0f;
          a1[h] = (lc == labA1) ? (_Float16)1.0f : (_Float16)0.0f;
        }
      }
#pragma unroll
      for (int h = 0; h < 16; ++h) {
        int krel = koffB / 2 + h + (h & 8);
        b0[h] = (stage[t0 * OVCHUNK + base + krel] == lb0) ? (_Float16)1.0f : (_Float16)0.0f;
        b1[h] = (stage[t1 * OVCHUNK + base + krel] == lb1) ? (_Float16)1.0f : (_Float16)0.0f;
      }
      acc00 = __builtin_amdgcn_wmma_f32_16x16x32_f16(false, a0, false, b0, (short)0, acc00, false, false);
      acc01 = __builtin_amdgcn_wmma_f32_16x16x32_f16(false, a0, false, b1, (short)0, acc01, false, false);
      acc10 = __builtin_amdgcn_wmma_f32_16x16x32_f16(false, a1, false, b0, (short)0, acc10, false, false);
      acc11 = __builtin_amdgcn_wmma_f32_16x16x32_f16(false, a1, false, b1, (short)0, acc11, false, false);
    }
    __syncthreads();
  }
  // D-tile layout: VGPR r -> row r (lanes 0-15) / row 8+r (lanes 16-31); col = lane&15
  float* o = OV + (size_t)slot * KCAP * KCAP;
#pragma unroll
  for (int r = 0; r < 8; ++r) {
    int jr = (hi ? 8 : 0) + r;
    int jr2 = 16 + jr;
    float vj = vval[slot * KCAP + jr];
    float vj2 = vval[slot * KCAP + jr2];
    float vk0 = ent_v[slot * KCAP + m];
    float vk1 = ent_v[slot * KCAP + 16 + m];
    o[jr * KCAP + m]       = acc00[r] * vj * vk0;
    o[jr * KCAP + 16 + m]  = acc01[r] * vj * vk1;
    o[jr2 * KCAP + m]      = acc10[r] * vj2 * vk0;
    o[jr2 * KCAP + 16 + m] = acc11[r] * vj2 * vk1;
  }
}

// ---------------------------------------------------------------- sequential matching scan
__global__ void k_scan(const float* __restrict__ OV, const float* __restrict__ sval,
                       const int* __restrict__ validc, const int* __restrict__ labs,
                       const float* __restrict__ vval, int* __restrict__ nbuf,
                       float* __restrict__ bar, int* __restrict__ ent_t,
                       int* __restrict__ ent_lab, float* __restrict__ ent_v,
                       int tcol, int step) {
  int slot = blockIdx.x;
  if (threadIdx.x != 0) return;
  int n = nbuf[slot];
  for (int j = 0; j < KCAP; ++j) {
    const float* row = OV + (size_t)slot * KCAP * KCAP + j * KCAP;
    float best = -1.f; int bi = 0;
    for (int k = 0; k < KCAP; ++k) {
      float f = row[k];
      if (f > best) { best = f; bi = k; }    // first max, like jnp.argmax
    }
    bool matched = best > 0.f;
    int widx = matched ? bi : n;
    widx = widx < 0 ? 0 : (widx > KCAP - 1 ? KCAP - 1 : widx);
    bool vj = validc[slot * KCAP + j] != 0;
    bool can = vj && (matched || n < KCAP);
    if (can) bar[(slot * KCAP + widx) * NSTEP + tcol] = sval[slot * KCAP + j];
    bool add = vj && !matched && (n < KCAP);
    if (add) {
      int nw = n < KCAP - 1 ? n : KCAP - 1;
      ent_t[slot * KCAP + nw] = step;
      ent_lab[slot * KCAP + nw] = labs[slot * KCAP + j];
      ent_v[slot * KCAP + nw] = vval[slot * KCAP + j];
      n += 1;
    }
  }
  nbuf[slot] = n;
}

// ---------------------------------------------------------------- final loss reduction
__global__ void k_loss(const float* __restrict__ bar, const int* __restrict__ nbuf,
                       float* __restrict__ out) {
  __shared__ float red[512];
  int tid = threadIdx.x;
  float acc = 0.f;
  if (tid < BCn * KCAP) {
    int slot = tid / KCAP, k = tid % KCAP;
    int np = nbuf[slot], nt = nbuf[BCn + slot];
    bool vp = k < np, ht = k < nt;
    const float* bp = bar + ((size_t)slot * KCAP + k) * NSTEP;
    const float* bt = bar + (size_t)BCn * KCAP * NSTEP + ((size_t)slot * KCAP + k) * NSTEP;
    float pmax = bp[0];
    for (int t = 1; t < NSTEP; ++t) pmax = bp[t] > pmax ? bp[t] : pmax;
    pmax = pmax > 0.f ? pmax : 1.f;
    float term = 0.f;
    for (int t = 0; t < NSTEP; ++t) {
      float tg = ht ? bt[t] : 0.f;
      float d = tg - bp[t];
      term += d * d;
    }
    acc = vp ? term / (pmax * pmax) : 0.f;
  }
  red[tid] = acc;
  __syncthreads();
  for (int o = 256; o > 0; o >>= 1) {
    if (tid < o) red[tid] += red[tid + o];
    __syncthreads();
  }
  if (tid == 0) out[0] = red[0] / (float)Bn;
}

// ---------------------------------------------------------------- launcher
extern "C" void kernel_launch(void* const* d_in, const int* in_sizes, int n_in,
                              void* d_out, int out_size, void* d_ws, size_t ws_size,
                              hipStream_t stream) {
  (void)in_sizes; (void)n_in; (void)out_size; (void)ws_size;
  const float* pred = (const float*)d_in[0];
  const float* gum  = (const float*)d_in[1];
  const int*   tgt  = (const int*)d_in[2];
  float* out = (float*)d_out;
  char* ws = (char*)d_ws;

  size_t off = 0;
  auto take = [&](size_t bytes) {
    size_t o = off;
    off = (off + bytes + 255) & ~(size_t)255;
    return o;
  };
  unsigned char* masks = (unsigned char*)(ws + take((size_t)NSTEP * BCn * HWn));        // 3.93 MB
  int*   labels = (int*)  (ws + take((size_t)NSTEP * BCn * HWn * 4));                   // 15.7 MB
  float* sizes  = (float*)(ws + take((size_t)BCn * HWn * 4));                           // 3.15 MB
  int*   maxL   = (int*)  (ws + take((size_t)BCn * 4));
  int*   labs   = (int*)  (ws + take((size_t)BCn * KCAP * 4));
  float* sval   = (float*)(ws + take((size_t)BCn * KCAP * 4));
  float* vval   = (float*)(ws + take((size_t)BCn * KCAP * 4));
  int*   validc = (int*)  (ws + take((size_t)BCn * KCAP * 4));
  int*   ent_t  = (int*)  (ws + take((size_t)BCn * KCAP * 4));
  int*   ent_lab= (int*)  (ws + take((size_t)BCn * KCAP * 4));
  float* ent_v  = (float*)(ws + take((size_t)BCn * KCAP * 4));
  int*   nbuf   = (int*)  (ws + take((size_t)2 * BCn * 4));
  float* bar    = (float*)(ws + take((size_t)2 * BCn * KCAP * NSTEP * 4));
  float* OV     = (float*)(ws + take((size_t)BCn * KCAP * KCAP * 4));
  // total workspace use: ~23 MB

  const int tvals[NSTEP] = {2, 3, 4, 1, 0};  // seq: binary, erode1, erode2, dil1, dil2
  const size_t MS = (size_t)BCn * HWn;       // one mask plane (bytes) / one label plane (elems)

  for (int src = 0; src < 2; ++src) {
    k_binarize<<<(Bn * HWn + 255) / 256, 256, 0, stream>>>(pred, gum, tgt, masks, src);
    k_morph<<<(BCn * HWn + 255) / 256, 256, 0, stream>>>(masks,          masks + 1 * MS, 0);
    k_morph<<<(BCn * HWn + 255) / 256, 256, 0, stream>>>(masks + 1 * MS, masks + 2 * MS, 0);
    k_morph<<<(BCn * HWn + 255) / 256, 256, 0, stream>>>(masks,          masks + 3 * MS, 1);
    k_morph<<<(BCn * HWn + 255) / 256, 256, 0, stream>>>(masks + 3 * MS, masks + 4 * MS, 1);
    k_regzero<<<(BCn * KCAP * NSTEP + 255) / 256, 256, 0, stream>>>(
        ent_t, ent_lab, ent_v, nbuf + src * BCn, bar + (size_t)src * BCn * KCAP * NSTEP);
    for (int s = 0; s < NSTEP; ++s) {
      k_cc<<<BCn, 1024, HWn * 4, stream>>>(masks + (size_t)s * MS, labels + (size_t)s * MS);
      k_sizes_zero<<<(BCn * HWn + 255) / 256, 256, 0, stream>>>(sizes, maxL);
      k_sizes_acc<<<(BCn * HWn + 255) / 256, 256, 0, stream>>>(labels + (size_t)s * MS, sizes, maxL);
      k_extract<<<BCn, 256, 0, stream>>>(labels + (size_t)s * MS, sizes, maxL,
                                         labs, sval, vval, validc);
      k_ov<<<BCn, 32, 0, stream>>>(labels, labs, vval, ent_t, ent_lab, ent_v, OV, s);
      k_scan<<<BCn, 32, 0, stream>>>(OV, sval, validc, labs, vval, nbuf + src * BCn,
                                     bar + (size_t)src * BCn * KCAP * NSTEP,
                                     ent_t, ent_lab, ent_v, tvals[s], s);
    }
  }
  k_loss<<<1, 512, 0, stream>>>(bar, nbuf, out);
}
